// MultiheadAttention_14559939133590
// MI455X (gfx1250) — matrix operand order repaired and checked by reading, not compiled, not run on hardware
//
#include <hip/hip_runtime.h>
#include <hip/hip_bf16.h>

// MultiheadAttention forward for MI455X (gfx1250, wave32, WMMA).
// B=2, T=2048, D=1024, H=16, HD=64.  All matmuls via v_wmma_f32_16x16x32_bf16.

#define DMODEL 1024
#define NHEAD  16
#define HDIM   64
#define SEQT   2048
#define BATCH  2
#define NROWS  (BATCH*SEQT)          // 4096 token rows

typedef __attribute__((ext_vector_type(16))) __bf16 bf16x16;
typedef __attribute__((ext_vector_type(8)))  __bf16 bf16x8;
typedef __attribute__((ext_vector_type(4)))  __bf16 bf16x4;
typedef __attribute__((ext_vector_type(8)))  float  f32x8;

static __device__ __forceinline__ __bf16 f2bf(float f) {
  unsigned u = __builtin_bit_cast(unsigned, f);
  unsigned r = u + 0x7FFFu + ((u >> 16) & 1u);     // round-to-nearest-even
  unsigned short s = (unsigned short)(r >> 16);
  return __builtin_bit_cast(__bf16, s);
}
static __device__ __forceinline__ float bf2f(__bf16 b) {
  unsigned short s = __builtin_bit_cast(unsigned short, b);
  unsigned u = ((unsigned)s) << 16;
  return __builtin_bit_cast(float, u);
}

union AB32 { bf16x16 f; bf16x8 h[2]; };   // one 16x32(A) or 32x16(B) bf16 fragment

static __device__ __forceinline__ f32x8 wmma_bf16(bf16x16 a, bf16x16 b, f32x8 c) {
  return __builtin_amdgcn_wmma_f32_16x16x32_bf16(
      /*neg_a=*/false, a, /*neg_b=*/false, b,
      /*c_mod=*/(short)0, c, /*reuse_a=*/false, /*reuse_b=*/false);
}

// ---------------------------------------------------------------- cast f32->bf16
__global__ __launch_bounds__(256) void cast_f32_bf16(const float* __restrict__ src,
                                                     __bf16* __restrict__ dst, int n4) {
  int i = blockIdx.x * 256 + threadIdx.x;
  if (i < n4) {
    float4 v = reinterpret_cast<const float4*>(src)[i];
    bf16x4 o;
    o[0] = f2bf(v.x); o[1] = f2bf(v.y); o[2] = f2bf(v.z); o[3] = f2bf(v.w);
    reinterpret_cast<bf16x4*>(dst)[i] = o;
  }
}

// ---------------------------------------------------------------- WMMA GEMM
// out[n, m] = sum_k A[n,k] * W[m,k]   (torch Linear: x @ W.T)
// A: (4096 x 1024) bf16 row-major; W: (1024 x 1024) bf16 row-major.
// Each wave computes a 64x64 tile as 4x4 WMMA sub-tiles. Block = 4 waves.
// mode 0: bf16 row-major out.  mode 1: f32 out + bias (final projection).
__global__ __launch_bounds__(128) void gemm_bf16_wmma(
    const __bf16* __restrict__ A, const __bf16* __restrict__ W,
    float* __restrict__ outF, __bf16* __restrict__ outB,
    const float* __restrict__ bias, int mode) {
  const int lane = threadIdx.x & 31;
  const int warp = threadIdx.x >> 5;
  const int lo   = lane & 15;
  const int half = lane >> 4;
  const int mBase = blockIdx.y * 64;
  const int nBase = blockIdx.x * 256 + warp * 64;
  const int Kd = DMODEL, Nn = DMODEL;

  f32x8 c[4][4];
#pragma unroll
  for (int mt = 0; mt < 4; ++mt)
#pragma unroll
    for (int nt = 0; nt < 4; ++nt) c[mt][nt] = (f32x8)0.0f;

  for (int kk = 0; kk < Kd; kk += 32) {
    if (kk + 32 < Kd) {   // warm next K-slab (global_prefetch_b8)
      __builtin_prefetch(A + (size_t)(mBase + lo) * Kd + kk + 32, 0, 3);
      __builtin_prefetch(W + (size_t)(nBase + lo) * Kd + kk + 32, 0, 3);
    }
    AB32 a[4];
#pragma unroll
    for (int mt = 0; mt < 4; ++mt) {
      const __bf16* p = A + (size_t)(mBase + mt * 16 + lo) * Kd + kk + half * 8;
      a[mt].h[0] = *reinterpret_cast<const bf16x8*>(p);        // K = +0..7
      a[mt].h[1] = *reinterpret_cast<const bf16x8*>(p + 16);   // K = +16..23
    }
#pragma unroll
    for (int nt = 0; nt < 4; ++nt) {
      AB32 b;
      const __bf16* p = W + (size_t)(nBase + nt * 16 + lo) * Kd + kk + half * 8;
      b.h[0] = *reinterpret_cast<const bf16x8*>(p);            // K = +0..7
      b.h[1] = *reinterpret_cast<const bf16x8*>(p + 16);       // K = +16..23
#pragma unroll
      for (int mt = 0; mt < 4; ++mt) c[mt][nt] = wmma_bf16(a[mt].f, b.f, c[mt][nt]);
    }
  }

#pragma unroll
  for (int mt = 0; mt < 4; ++mt)
#pragma unroll
    for (int nt = 0; nt < 4; ++nt) {
      const int col = nBase + nt * 16 + lo;
#pragma unroll
      for (int r = 0; r < 8; ++r) {
        const int row = mBase + mt * 16 + r + half * 8;   // D-layout: M = r + 8*half
        float v = c[mt][nt][r];
        if (mode == 1) outF[(size_t)row * Nn + col] = v + bias[col];
        else           outB[(size_t)row * Nn + col] = f2bf(v);
      }
    }
}

// ---------------------------------------------------------------- RoPE + rearrange
// z=0: q proj -> RoPE -> Qh (b,h,t,64);  z=1: k -> RoPE -> Kh;  z=2: v -> Vt (b,h,64,t)
__global__ __launch_bounds__(256) void rope_rearrange(
    const __bf16* __restrict__ pq, const __bf16* __restrict__ pk,
    const __bf16* __restrict__ pv,
    __bf16* __restrict__ Qh, __bf16* __restrict__ Kh, __bf16* __restrict__ Vt) {
  const int z = blockIdx.z;
  const int p = blockIdx.x * 256 + threadIdx.x;       // pair index, < B*T*D/2
  const int dh = p % (DMODEL / 2);
  const int t  = (p / (DMODEL / 2)) % SEQT;
  const int b  = p / ((DMODEL / 2) * SEQT);
  const int d0 = dh * 2;
  const int h  = d0 >> 6;           // / HDIM
  const int hd0 = d0 & 63;          // % HDIM
  const size_t srcIdx = ((size_t)(b * SEQT + t)) * DMODEL + d0;

  if (z < 2) {
    const __bf16* src = (z == 0) ? pq : pk;
    float x0 = bf2f(src[srcIdx]);
    float x1 = bf2f(src[srcIdx + 1]);
    const int j = hd0 >> 1;
    // theta_j = 10000^(-2j/64);  ln(10000) = 9.210340371976184
    float freq = __expf(-(float)(2 * j) * (9.210340371976184f / 64.0f));
    float ang = (float)t * freq, sn, cs;
    __sincosf(ang, &sn, &cs);
    float o0 = x0 * cs - x1 * sn;
    float o1 = x1 * cs + x0 * sn;
    __bf16* dst = (z == 0) ? Qh : Kh;
    size_t base = (((size_t)(b * NHEAD + h)) * SEQT + t) * HDIM + hd0;
    dst[base]     = f2bf(o0);
    dst[base + 1] = f2bf(o1);
  } else {
    float v0 = bf2f(pv[srcIdx]);
    float v1 = bf2f(pv[srcIdx + 1]);
    size_t base = (((size_t)(b * NHEAD + h)) * HDIM + hd0) * SEQT + t;  // transposed
    Vt[base]        = f2bf(v0);
    Vt[base + SEQT] = f2bf(v1);
  }
}

// ---------------------------------------------------------------- flash attention
// One wave per 16-row Q tile. Online softmax; scores and P@V via WMMA.
// Q/K head-major (bh, t, 64); Vt head-major transposed (bh, 64, t).
__global__ __launch_bounds__(256) void flash_attn(
    const __bf16* __restrict__ Q, const __bf16* __restrict__ K,
    const __bf16* __restrict__ Vt, __bf16* __restrict__ AO) {
  __shared__ __attribute__((aligned(16))) __bf16 ldsP[8][16][40];  // per-wave 16x32 (+pad)

  const int lane = threadIdx.x & 31;
  const int warp = threadIdx.x >> 5;
  const int lo   = lane & 15;
  const int half = lane >> 4;
  const int tile = blockIdx.x * 8 + warp;   // 0..4095
  const int bh = tile >> 7;                 // 128 q-tiles per (b,h)
  const int qt = tile & 127;
  const int b = bh >> 4, h = bh & 15;

  const __bf16* Qb = Q  + (((size_t)bh * SEQT) + qt * 16) * HDIM;
  const __bf16* Kb = K  + ((size_t)bh * SEQT) * HDIM;
  const __bf16* Vb = Vt + ((size_t)bh * HDIM) * SEQT;

  AB32 qa[2];
#pragma unroll
  for (int cc = 0; cc < 2; ++cc) {
    const __bf16* p = Qb + lo * HDIM + cc * 32 + half * 8;
    qa[cc].h[0] = *reinterpret_cast<const bf16x8*>(p);
    qa[cc].h[1] = *reinterpret_cast<const bf16x8*>(p + 16);
  }

  f32x8 o[4];
#pragma unroll
  for (int nt = 0; nt < 4; ++nt) o[nt] = (f32x8)0.0f;
  float rmax[8], rsum[8];
#pragma unroll
  for (int r = 0; r < 8; ++r) { rmax[r] = -1e30f; rsum[r] = 0.0f; }

  for (int kb = 0; kb < SEQT; kb += 32) {
    // ---- scores S = Q @ K^T for 32 keys (two 16x16 tiles), K-dim = HD = 64
    f32x8 s[2];
#pragma unroll
    for (int n = 0; n < 2; ++n) {
      f32x8 acc = (f32x8)0.0f;
#pragma unroll
      for (int cc = 0; cc < 2; ++cc) {
        AB32 kf;
        const __bf16* kp = Kb + (size_t)(kb + n * 16 + lo) * HDIM + cc * 32 + half * 8;
        kf.h[0] = *reinterpret_cast<const bf16x8*>(kp);
        kf.h[1] = *reinterpret_cast<const bf16x8*>(kp + 16);
        acc = wmma_bf16(qa[cc].f, kf.f, acc);
      }
      s[n] = acc;
    }

    // ---- online softmax (rows m = r + 8*half; reduce across 16 lanes of the half)
#pragma unroll
    for (int r = 0; r < 8; ++r) {
      float a0 = s[0][r] * 0.125f;           // 1/sqrt(64)
      float a1 = s[1][r] * 0.125f;
      float mx = fmaxf(a0, a1);
      mx = fmaxf(mx, __shfl_xor(mx, 1, 32));
      mx = fmaxf(mx, __shfl_xor(mx, 2, 32));
      mx = fmaxf(mx, __shfl_xor(mx, 4, 32));
      mx = fmaxf(mx, __shfl_xor(mx, 8, 32));
      float nm = fmaxf(rmax[r], mx);
      float corr = __expf(rmax[r] - nm);
      float p0 = __expf(a0 - nm);
      float p1 = __expf(a1 - nm);
      float sm = p0 + p1;
      sm += __shfl_xor(sm, 1, 32);
      sm += __shfl_xor(sm, 2, 32);
      sm += __shfl_xor(sm, 4, 32);
      sm += __shfl_xor(sm, 8, 32);
      rsum[r] = rsum[r] * corr + sm;
      rmax[r] = nm;
#pragma unroll
      for (int nt = 0; nt < 4; ++nt) o[nt][r] *= corr;
      const int m = r + half * 8;
      ldsP[warp][m][lo]      = f2bf(p0);     // C-layout -> A-layout via LDS
      ldsP[warp][m][16 + lo] = f2bf(p1);
    }

    // ---- reload P as a 16x32 A-fragment (same-wave DS ops are in-order)
    AB32 pa;
    pa.h[0] = *reinterpret_cast<const bf16x8*>(&ldsP[warp][lo][half * 8]);
    pa.h[1] = *reinterpret_cast<const bf16x8*>(&ldsP[warp][lo][half * 8 + 16]);

    // ---- O += P @ V  (B-frag columns = rows of transposed V, contiguous in t)
#pragma unroll
    for (int nt = 0; nt < 4; ++nt) {
      AB32 vf;
      const __bf16* vp = Vb + (size_t)(nt * 16 + lo) * SEQT + kb + half * 8;
      vf.h[0] = *reinterpret_cast<const bf16x8*>(vp);
      vf.h[1] = *reinterpret_cast<const bf16x8*>(vp + 16);
      o[nt] = wmma_bf16(pa.f, vf.f, o[nt]);
    }
  }

  // ---- normalize and write attention output (row-major (B*T, D) bf16)
#pragma unroll
  for (int nt = 0; nt < 4; ++nt)
#pragma unroll
    for (int r = 0; r < 8; ++r) {
      float v = o[nt][r] / rsum[r];
      size_t row = (size_t)b * SEQT + qt * 16 + r + half * 8;
      size_t col = (size_t)h * HDIM + nt * 16 + lo;
      AO[row * DMODEL + col] = f2bf(v);
    }
}

// ---------------------------------------------------------------- host launcher
extern "C" void kernel_launch(void* const* d_in, const int* in_sizes, int n_in,
                              void* d_out, int out_size, void* d_ws, size_t ws_size,
                              hipStream_t stream) {
  const float* query = (const float*)d_in[0];
  const float* key   = (const float*)d_in[1];
  const float* value = (const float*)d_in[2];
  const float* Wq = (const float*)d_in[3];
  const float* Wk = (const float*)d_in[5];
  const float* Wv = (const float*)d_in[7];
  const float* Wo = (const float*)d_in[9];
  const float* bo = (const float*)d_in[10];

  const size_t ND = (size_t)NROWS * DMODEL;   // 4M elems
  const size_t DD = (size_t)DMODEL * DMODEL;  // 1M elems
  char* ws = (char*)d_ws;

  // workspace (bf16): 3*ND (acts) + 4*DD (weights) + 3*ND (projections) = 56 MB
  __bf16* xq = (__bf16*)ws;                 // later reused as Qh (head-major)
  __bf16* xk = xq + ND;                     // later reused as Kh
  __bf16* xv = xk + ND;                     // later reused as Vt (transposed)
  __bf16* wq = xv + ND;
  __bf16* wk = wq + DD;
  __bf16* wv = wk + DD;
  __bf16* wo = wv + DD;
  __bf16* pq = wo + DD;                     // later reused as attention output
  __bf16* pk = pq + ND;
  __bf16* pv = pk + ND;

  // 1) casts to bf16
  {
    int n4a = (int)(ND / 4), n4w = (int)(DD / 4);
    dim3 blk(256);
    cast_f32_bf16<<<dim3((n4a + 255) / 256), blk, 0, stream>>>(query, xq, n4a);
    cast_f32_bf16<<<dim3((n4a + 255) / 256), blk, 0, stream>>>(key,   xk, n4a);
    cast_f32_bf16<<<dim3((n4a + 255) / 256), blk, 0, stream>>>(value, xv, n4a);
    cast_f32_bf16<<<dim3((n4w + 255) / 256), blk, 0, stream>>>(Wq, wq, n4w);
    cast_f32_bf16<<<dim3((n4w + 255) / 256), blk, 0, stream>>>(Wk, wk, n4w);
    cast_f32_bf16<<<dim3((n4w + 255) / 256), blk, 0, stream>>>(Wv, wv, n4w);
    cast_f32_bf16<<<dim3((n4w + 255) / 256), blk, 0, stream>>>(Wo, wo, n4w);
  }

  // 2) Q/K/V projections (bf16 out)
  dim3 ggrid(DMODEL / 256, NROWS / 64), gblk(128);
  gemm_bf16_wmma<<<ggrid, gblk, 0, stream>>>(xq, wq, nullptr, pq, nullptr, 0);
  gemm_bf16_wmma<<<ggrid, gblk, 0, stream>>>(xk, wk, nullptr, pk, nullptr, 0);
  gemm_bf16_wmma<<<ggrid, gblk, 0, stream>>>(xv, wv, nullptr, pv, nullptr, 0);

  // 3) RoPE + head-major rearrange (writes over xq/xk/xv, which are now free)
  {
    int pairs = (int)(ND / 2);
    rope_rearrange<<<dim3(pairs / 256, 1, 3), dim3(256), 0, stream>>>(
        pq, pk, pv, /*Qh=*/xq, /*Kh=*/xk, /*Vt=*/xv);
  }

  // 4) flash attention: 4096 q-tiles, 8 waves/block
  flash_attn<<<dim3(4096 / 8), dim3(256), 0, stream>>>(xq, xk, xv, /*AO=*/pq);

  // 5) output projection: f32 out + bias
  gemm_bf16_wmma<<<ggrid, gblk, 0, stream>>>(pq, wo, (float*)d_out, nullptr, bo, 1);
}
